// Model_44504451121624
// MI455X (gfx1250) — compile-verified, matrix operand order checked by reading, not run on hardware
//
#include <hip/hip_runtime.h>
#include <stdint.h>

// ---- problem constants (match reference) ----
#define NN      512      // nodes
#define UNITS   64
#define MDIFF   5        // diffusion matrices
#define T_INS   12
#define T_OUTS  12
#define BATCH   128
#define NEDGE   8192
#define TOK     (NN * BATCH)   // 65536 tokens (n-major, b-minor)

typedef unsigned short bf16_t;
typedef __bf16 bf16x16 __attribute__((ext_vector_type(16)));
typedef float  f32x8   __attribute__((ext_vector_type(8)));
typedef int    v4i_g   __attribute__((vector_size(4 * sizeof(int))));  // matches builtin param

#if __has_builtin(__builtin_amdgcn_global_load_async_to_lds_b128)
#define ASYNC_LDS 1
#else
#define ASYNC_LDS 0
#endif

__device__ __forceinline__ void wait_async_lds() {
#if __has_builtin(__builtin_amdgcn_s_wait_asynccnt)
  __builtin_amdgcn_s_wait_asynccnt(0);
#else
  asm volatile("s_wait_asynccnt 0x0" ::: "memory");
#endif
}

__device__ __forceinline__ bf16_t f2bf(float f) {
  union { float f; uint32_t u; } v; v.f = f;
  uint32_t r = v.u + 0x7FFFu + ((v.u >> 16) & 1u);
  return (bf16_t)(r >> 16);
}
__device__ __forceinline__ float bf2f(bf16_t h) {
  union { uint32_t u; float f; } v; v.u = ((uint32_t)h) << 16;
  return v.f;
}

// ---------------- utility kernels ----------------
__global__ void fill_zero_f32(float* __restrict__ p, int n) {
  int i = blockIdx.x * blockDim.x + threadIdx.x;
  if (i < n) p[i] = 0.0f;
}

__global__ void scatter_edges(const int* __restrict__ ei,
                              const float* __restrict__ ew,
                              float* __restrict__ A) {
  int e = blockIdx.x * blockDim.x + threadIdx.x;
  if (e >= NEDGE) return;
  int s = ei[e], d = ei[NEDGE + e];
  atomicAdd(&A[(size_t)s * NN + d], ew[e]);
}

__global__ void degrees_recip(const float* __restrict__ A,
                              float* __restrict__ rrow, float* __restrict__ rcol) {
  int i = blockIdx.x * blockDim.x + threadIdx.x;
  if (i >= NN) return;
  float sr = 0.f, sc = 0.f;
  for (int j = 0; j < NN; ++j) { sr += A[(size_t)i * NN + j]; sc += A[(size_t)j * NN + i]; }
  rrow[i] = (sr > 0.f) ? 1.f / sr : 0.f;
  rcol[i] = (sc > 0.f) ? 1.f / sc : 0.f;
}

// S0 = rw(A) = (D^-1 A)^T ; S1 = rw(A^T)
__global__ void make_supports(const float* __restrict__ A,
                              const float* __restrict__ rrow,
                              const float* __restrict__ rcol,
                              bf16_t* __restrict__ S0, bf16_t* __restrict__ S1) {
  int id = blockIdx.x * blockDim.x + threadIdx.x;
  if (id >= NN * NN) return;
  int j = id % NN, i = id / NN;
  S0[id] = f2bf(A[(size_t)j * NN + i] * rrow[j]);
  S1[id] = f2bf(A[(size_t)i * NN + j] * rcol[j]);
}

// reorder W rows (c*M+m) -> (m*C+c), convert to bf16
__global__ void permute_w(const float* __restrict__ W, bf16_t* __restrict__ Wp,
                          int C, int OUT) {
  int id = blockIdx.x * blockDim.x + threadIdx.x;
  int total = MDIFF * C * OUT;
  if (id >= total) return;
  int o = id % OUT, k = id / OUT;
  int m = k / C, c = k % C;
  Wp[id] = f2bf(W[((size_t)c * MDIFF + m) * OUT + o]);
}

// build X0 block: [n][b][c] bf16 from x([B,N] or [B,N*64]) and h([B,N*64])
__global__ void concat_xh(const float* __restrict__ X, int dimX,
                          const float* __restrict__ H,
                          bf16_t* __restrict__ X0, int C) {
  int id = blockIdx.x * blockDim.x + threadIdx.x;
  int total = TOK * C;
  if (id >= total) return;
  int c = id % C, t = id / C;
  int b = t % BATCH, n = t / BATCH;
  float v;
  if (c < dimX)
    v = (dimX == 1) ? X[(size_t)b * NN + n]
                    : X[((size_t)b * NN + n) * UNITS + c];
  else
    v = H[((size_t)b * NN + n) * UNITS + (c - dimX)];
  X0[id] = f2bf(v);
}

// rh = r (gate cols 0..63) * h
__global__ void compute_rh(const float* __restrict__ gates,
                           const float* __restrict__ h, float* __restrict__ rh) {
  int id = blockIdx.x * blockDim.x + threadIdx.x;
  if (id >= TOK * UNITS) return;
  int u = id % UNITS, bn = id / UNITS;
  int n = bn % NN, b = bn / NN;
  int tok = n * BATCH + b;
  rh[id] = gates[(size_t)tok * 128 + u] * h[id];
}

// h' = u*h + (1-u)*c
__global__ void gru_update(const float* __restrict__ gates,
                           const float* __restrict__ cand,
                           const float* __restrict__ h, float* __restrict__ hn) {
  int id = blockIdx.x * blockDim.x + threadIdx.x;
  if (id >= TOK * UNITS) return;
  int u = id % UNITS, bn = id / UNITS;
  int n = bn % NN, b = bn / NN;
  int tok = n * BATCH + b;
  float U = gates[(size_t)tok * 128 + 64 + u];
  hn[id] = U * h[id] + (1.f - U) * cand[(size_t)tok * UNITS + u];
}

// out[b][step][n] = h1[b][n][:] . projW + projb ; also next decoder input
__global__ void dec_project(const float* __restrict__ h1,
                            const float* __restrict__ pW, const float* __restrict__ pb,
                            float* __restrict__ dout, float* __restrict__ nextin,
                            int step) {
  int id = blockIdx.x * blockDim.x + threadIdx.x;
  if (id >= BATCH * NN) return;
  int n = id % NN, b = id / NN;
  const float* hr = h1 + ((size_t)b * NN + n) * UNITS;
  float s = pb[0];
  #pragma unroll 8
  for (int u = 0; u < UNITS; ++u) s += hr[u] * pW[u];
  dout[((size_t)b * T_OUTS + step) * NN + n] = s;
  nextin[id] = s;
}

// ---------------- WMMA GEMM kernels ----------------
// LDS row pitch in halfs: 80B rows -> 16B aligned for b128, 20-bank stride
#define LDT 40

// Y[N x F] = alpha * (S[N x N] @ X[N x F]) + beta * Z[N x F]  (bf16, f32 accum)
// double-buffered LDS, async A-tile DMA (or reg-pipelined fallback), pipelined B tile
__global__ __launch_bounds__(256)
void spmm_wmma(const bf16_t* __restrict__ S, const bf16_t* __restrict__ X,
               bf16_t* __restrict__ Y, const bf16_t* __restrict__ Z,
               int F, float alpha, float beta) {
  __shared__ bf16_t As[2][128 * LDT];
  __shared__ bf16_t Bts[2][64 * LDT];
  const int tid  = threadIdx.x;
  const int wave = tid >> 5, lane = tid & 31;
  const int half = lane >> 4, r = lane & 15;
  const int col0 = blockIdx.x * 64;
  const int row0 = blockIdx.y * 128;

  // B staging: one uint4 (8 halfs along columns) per thread per tile
  const int bk  = tid >> 3;          // k row 0..31
  const int bc0 = (tid & 7) * 8;     // col start
  // A staging: two b128 (8 halfs) per thread per tile (128 rows x 4 segs)
  const int ar0 = tid >> 2,          as0 = (tid & 3) * 8;
  const int ar1 = (tid + 256) >> 2,  as1 = ((tid + 256) & 3) * 8;

  f32x8 acc[4];
  #pragma unroll
  for (int j = 0; j < 4; ++j)
    #pragma unroll
    for (int v = 0; v < 8; ++v) acc[j][v] = 0.0f;

#if !ASYNC_LDS
  uint4 areg0, areg1;
#endif
  uint4 breg;

  auto issueA = [&](int buf, int kk) {
#if ASYNC_LDS
    __builtin_amdgcn_global_load_async_to_lds_b128(
        (v4i_g*)&S[(size_t)(row0 + ar0) * NN + kk + as0],
        (v4i_g*)&As[buf][ar0 * LDT + as0], 0, 0);
    __builtin_amdgcn_global_load_async_to_lds_b128(
        (v4i_g*)&S[(size_t)(row0 + ar1) * NN + kk + as1],
        (v4i_g*)&As[buf][ar1 * LDT + as1], 0, 0);
#else
    areg0 = *(const uint4*)&S[(size_t)(row0 + ar0) * NN + kk + as0];
    areg1 = *(const uint4*)&S[(size_t)(row0 + ar1) * NN + kk + as1];
#endif
  };
  auto loadB = [&](int kk) {
    breg = *(const uint4*)&X[(size_t)(kk + bk) * F + col0 + bc0];
  };
  auto commit = [&](int buf) {
#if ASYNC_LDS
    wait_async_lds();
#else
    *(uint4*)&As[buf][ar0 * LDT + as0] = areg0;
    *(uint4*)&As[buf][ar1 * LDT + as1] = areg1;
#endif
    const bf16_t* bh = (const bf16_t*)&breg;
    #pragma unroll
    for (int i = 0; i < 8; ++i)
      Bts[buf][(bc0 + i) * LDT + bk] = bh[i];
  };
  auto compute = [&](int buf) {
    union Frag { bf16x16 v; uint32_t u[8]; } a, b;
    const uint32_t* arow = (const uint32_t*)&As[buf][(wave * 16 + r) * LDT];
    #pragma unroll
    for (int g = 0; g < 2; ++g)
      #pragma unroll
      for (int j = 0; j < 4; ++j)
        a.u[g * 4 + j] = arow[g * 8 + half * 4 + j];
    #pragma unroll
    for (int jt = 0; jt < 4; ++jt) {
      const uint32_t* brow = (const uint32_t*)&Bts[buf][(jt * 16 + r) * LDT];
      #pragma unroll
      for (int g = 0; g < 2; ++g)
        #pragma unroll
        for (int j = 0; j < 4; ++j)
          b.u[g * 4 + j] = brow[g * 8 + half * 4 + j];
      acc[jt] = __builtin_amdgcn_wmma_f32_16x16x32_bf16(
          false, a.v, false, b.v, (short)0, acc[jt], false, false);
    }
  };

  // prologue: stage tile 0
  issueA(0, 0); loadB(0); commit(0);
  __syncthreads();

  for (int kk = 0; kk < NN; kk += 32) {
    const int cur = (kk >> 5) & 1, nxt = cur ^ 1;
    const bool more = (kk + 32) < NN;
    if (more) { issueA(nxt, kk + 32); loadB(kk + 32); }  // loads in flight
    compute(cur);                                        // WMMA overlaps loads
    if (more) { commit(nxt); __syncthreads(); }
  }

  #pragma unroll
  for (int jt = 0; jt < 4; ++jt)
    #pragma unroll
    for (int v = 0; v < 8; ++v) {
      int row = row0 + wave * 16 + v + 8 * half;
      int col = col0 + jt * 16 + r;
      size_t idx = (size_t)row * F + col;
      float val = alpha * acc[jt][v];
      if (beta != 0.0f) val += beta * bf2f(Z[idx]);
      Y[idx] = f2bf(val);
    }
}

// out[T x OUT] = act( Xall(5 blocks of [T x C]) @ Wp[5C x OUT] + bias )
__global__ __launch_bounds__(256)
void proj_wmma(const bf16_t* __restrict__ Xall, const bf16_t* __restrict__ Wp,
               const float* __restrict__ bias, float* __restrict__ out,
               int C, int OUT, int act) {
  __shared__ bf16_t As[128 * LDT];
  __shared__ bf16_t Bts[64 * LDT];
  const int tid  = threadIdx.x;
  const int wave = tid >> 5, lane = tid & 31;
  const int half = lane >> 4, r = lane & 15;
  const int o0 = blockIdx.x * 64;
  const int t0 = blockIdx.y * 128;

  f32x8 acc[4];
  #pragma unroll
  for (int j = 0; j < 4; ++j)
    #pragma unroll
    for (int v = 0; v < 8; ++v) acc[j][v] = 0.0f;

  const bool cvec = ((C & 7) == 0);

  for (int m = 0; m < MDIFF; ++m) {
    const bf16_t* Xm = Xall + (size_t)m * TOK * C;
    for (int kc = 0; kc < C; kc += 32) {
      // ---- A tile ----
      if (cvec) {
        #pragma unroll
        for (int t = tid; t < 128 * 4; t += 256) {   // 512 uint4
          int row = t >> 2, seg = (t & 3) * 8;
          *(uint4*)&As[row * LDT + seg] =
              *(const uint4*)&Xm[(size_t)(t0 + row) * C + kc + seg];
        }
      } else {
        for (int t = tid; t < 128 * 32; t += 256) {
          int row = t >> 5, k = t & 31, kg = kc + k;
          As[row * LDT + k] = (kg < C) ? Xm[(size_t)(t0 + row) * C + kg] : (bf16_t)0;
        }
      }
      // ---- B tile (weights), one predicated uint4 per thread ----
      {
        int k = tid >> 3, kg = kc + k, o = (tid & 7) * 8;
        uint4 d; d.x = d.y = d.z = d.w = 0u;
        if (kg < C)
          d = *(const uint4*)&Wp[((size_t)m * C + kg) * OUT + o0 + o];
        const bf16_t* dh = (const bf16_t*)&d;
        #pragma unroll
        for (int i = 0; i < 8; ++i)
          Bts[(o + i) * LDT + k] = dh[i];
      }
      if (kc + 32 < C)
        __builtin_prefetch(&Xm[(size_t)(t0 + (tid >> 1)) * C + kc + 32], 0, 1);
      __syncthreads();

      union Frag { bf16x16 v; uint32_t u[8]; } a, b;
      const uint32_t* arow = (const uint32_t*)&As[(wave * 16 + r) * LDT];
      #pragma unroll
      for (int g = 0; g < 2; ++g)
        #pragma unroll
        for (int j = 0; j < 4; ++j)
          a.u[g * 4 + j] = arow[g * 8 + half * 4 + j];
      #pragma unroll
      for (int jt = 0; jt < 4; ++jt) {
        const uint32_t* brow = (const uint32_t*)&Bts[(jt * 16 + r) * LDT];
        #pragma unroll
        for (int g = 0; g < 2; ++g)
          #pragma unroll
          for (int j = 0; j < 4; ++j)
            b.u[g * 4 + j] = brow[g * 8 + half * 4 + j];
        acc[jt] = __builtin_amdgcn_wmma_f32_16x16x32_bf16(
            false, a.v, false, b.v, (short)0, acc[jt], false, false);
      }
      __syncthreads();
    }
  }

  #pragma unroll
  for (int jt = 0; jt < 4; ++jt)
    #pragma unroll
    for (int v = 0; v < 8; ++v) {
      int row = t0 + wave * 16 + v + 8 * half;
      int col = o0 + jt * 16 + r;
      float val = acc[jt][v] + bias[col];
      if (act == 1)      val = 1.0f / (1.0f + __expf(-val));
      else if (act == 2) val = tanhf(val);
      out[(size_t)row * OUT + col] = val;
    }
}

// ---------------- host orchestration ----------------
extern "C" void kernel_launch(void* const* d_in, const int* in_sizes, int n_in,
                              void* d_out, int out_size, void* d_ws, size_t ws_size,
                              hipStream_t stream) {
  (void)in_sizes; (void)n_in; (void)out_size; (void)ws_size;
  const float* hist  = (const float*)d_in[0];
  const int*   eidx  = (const int*)d_in[1];
  const float* eattr = (const float*)d_in[2];
  struct Layer { const float *Wg, *bg, *Wc, *bc; };
  Layer L[4]; // enc0, enc1, dec0, dec1
  for (int i = 0; i < 4; ++i) {
    L[i].Wg = (const float*)d_in[3 + i * 4 + 0];
    L[i].bg = (const float*)d_in[3 + i * 4 + 1];
    L[i].Wc = (const float*)d_in[3 + i * 4 + 2];
    L[i].bc = (const float*)d_in[3 + i * 4 + 3];
  }
  const float* projW = (const float*)d_in[19];
  const float* projB = (const float*)d_in[20];
  float* out = (float*)d_out;

  // workspace carve-up
  char* ws = (char*)d_ws;
  size_t off = 0;
  auto alloc = [&](size_t bytes) -> void* {
    void* p = ws + off;
    off += (bytes + 255) & ~(size_t)255;
    return p;
  };
  float*  A     = (float*)alloc((size_t)NN * NN * 4);
  float*  rrow  = (float*)alloc(NN * 4);
  float*  rcol  = (float*)alloc(NN * 4);
  bf16_t* S0    = (bf16_t*)alloc((size_t)NN * NN * 2);
  bf16_t* S1    = (bf16_t*)alloc((size_t)NN * NN * 2);
  int  Cl[4]  = {65, 128, 65, 128};
  bf16_t* Wpg[4]; bf16_t* Wpc[4];
  for (int i = 0; i < 4; ++i) {
    Wpg[i] = (bf16_t*)alloc((size_t)MDIFF * Cl[i] * 128 * 2);
    Wpc[i] = (bf16_t*)alloc((size_t)MDIFF * Cl[i] * 64 * 2);
  }
  bf16_t* Xall  = (bf16_t*)alloc((size_t)MDIFF * TOK * 128 * 2);
  float*  gates = (float*)alloc((size_t)TOK * 128 * 4);
  float*  cand  = (float*)alloc((size_t)TOK * 64 * 4);
  float*  rh    = (float*)alloc((size_t)TOK * 64 * 4);
  float*  hb[4];
  for (int i = 0; i < 4; ++i) hb[i] = (float*)alloc((size_t)TOK * UNITS * 4);
  float*  inA = (float*)alloc((size_t)BATCH * NN * 4);
  float*  inB = (float*)alloc((size_t)BATCH * NN * 4);

  auto cdiv = [](int a, int b) { return (a + b - 1) / b; };

  // ---- supports ----
  fill_zero_f32<<<cdiv(NN * NN, 256), 256, 0, stream>>>(A, NN * NN);
  scatter_edges<<<cdiv(NEDGE, 256), 256, 0, stream>>>(eidx, eattr, A);
  degrees_recip<<<cdiv(NN, 256), 256, 0, stream>>>(A, rrow, rcol);
  make_supports<<<cdiv(NN * NN, 256), 256, 0, stream>>>(A, rrow, rcol, S0, S1);

  // ---- weight permute/convert ----
  for (int i = 0; i < 4; ++i) {
    int tg = MDIFF * Cl[i] * 128, tc = MDIFF * Cl[i] * 64;
    permute_w<<<cdiv(tg, 256), 256, 0, stream>>>(L[i].Wg, Wpg[i], Cl[i], 128);
    permute_w<<<cdiv(tc, 256), 256, 0, stream>>>(L[i].Wc, Wpc[i], Cl[i], 64);
  }

  // ---- gconv: build Xall, 4 diffusion SpMMs, projection GEMM ----
  auto gconv = [&](const float* x, int dimX, const float* hvec,
                   const bf16_t* Wp, const float* bias, int OUT, int act,
                   float* obuf) {
    int C = dimX + UNITS;
    int F = BATCH * C;
    size_t blk = (size_t)TOK * C;
    concat_xh<<<cdiv(TOK * C, 256), 256, 0, stream>>>(x, dimX, hvec, Xall, C);
    dim3 g(F / 64, NN / 128);
    spmm_wmma<<<g, 256, 0, stream>>>(S0, Xall,           Xall + blk,     nullptr, F, 1.f,  0.f);
    spmm_wmma<<<g, 256, 0, stream>>>(S0, Xall + blk,     Xall + 2 * blk, Xall,    F, 2.f, -1.f);
    spmm_wmma<<<g, 256, 0, stream>>>(S1, Xall,           Xall + 3 * blk, nullptr, F, 1.f,  0.f);
    spmm_wmma<<<g, 256, 0, stream>>>(S1, Xall + 3 * blk, Xall + 4 * blk, Xall,    F, 2.f, -1.f);
    dim3 gp(OUT / 64, TOK / 128);
    proj_wmma<<<gp, 256, 0, stream>>>(Xall, Wp, bias, obuf, C, OUT, act);
  };

  auto cell = [&](const float* x, int dimX, float* hcur, int li, float* hnew) {
    gconv(x, dimX, hcur, Wpg[li], L[li].bg, 128, /*sigmoid*/1, gates);
    compute_rh<<<cdiv(TOK * UNITS, 256), 256, 0, stream>>>(gates, hcur, rh);
    gconv(x, dimX, rh, Wpc[li], L[li].bc, 64, /*tanh*/2, cand);
    gru_update<<<cdiv(TOK * UNITS, 256), 256, 0, stream>>>(gates, cand, hcur, hnew);
  };

  // ---- encoder ----
  float *h0c = hb[0], *h0n = hb[1], *h1c = hb[2], *h1n = hb[3];
  fill_zero_f32<<<cdiv(TOK * UNITS, 256), 256, 0, stream>>>(h0c, TOK * UNITS);
  fill_zero_f32<<<cdiv(TOK * UNITS, 256), 256, 0, stream>>>(h1c, TOK * UNITS);
  for (int t = 0; t < T_INS; ++t) {
    cell(hist + (size_t)t * BATCH * NN, 1, h0c, 0, h0n);
    cell(h0n, UNITS, h1c, 1, h1n);
    float* tmp;
    tmp = h0c; h0c = h0n; h0n = tmp;
    tmp = h1c; h1c = h1n; h1n = tmp;
  }

  // ---- decoder ----
  fill_zero_f32<<<cdiv(BATCH * NN, 256), 256, 0, stream>>>(inA, BATCH * NN);
  float *incur = inA, *innext = inB;
  for (int s = 0; s < T_OUTS; ++s) {
    cell(incur, 1, h0c, 2, h0n);
    cell(h0n, UNITS, h1c, 3, h1n);
    dec_project<<<cdiv(BATCH * NN, 256), 256, 0, stream>>>(h1n, projW, projB, out, innext, s);
    float* tmp;
    tmp = h0c; h0c = h0n; h0n = tmp;
    tmp = h1c; h1c = h1n; h1n = tmp;
    tmp = incur; incur = innext; innext = tmp;
  }
}
